// MultiHeadAttention_54760833024425
// MI455X (gfx1250) — matrix operand order repaired and checked by reading, not compiled, not run on hardware
//
#include <hip/hip_runtime.h>
#include <hip/hip_bf16.h>
#include <math.h>
#include <stdint.h>

typedef __attribute__((ext_vector_type(16))) _Float16 v16h;
typedef __attribute__((ext_vector_type(8)))  _Float16 v8h;
typedef __attribute__((ext_vector_type(8)))  float    v8f;

#define DIM     2048
#define S_LEN   2048
#define NH      32
#define NKV     8
#define HD      64
#define KVD     (NKV*HD)   // 512

union Frag { v16h v; v8h h[2]; };

// CDNA5 async copy: global -> LDS, per-lane 16B, tracked by ASYNCcnt.
__device__ __forceinline__ void async_ld_b128(unsigned lds_off, const void* g) {
  asm volatile("global_load_async_to_lds_b128 %0, %1, off"
               :: "v"(lds_off), "v"((unsigned long long)(uintptr_t)g)
               : "memory");
}
__device__ __forceinline__ void wait_async0() {
  asm volatile("s_wait_asynccnt 0" ::: "memory");
}

// ---------------------------------------------------------------- f32 -> f16
__global__ void f32_to_f16_kernel(const float* __restrict__ in,
                                  _Float16* __restrict__ out, int n) {
  int i = blockIdx.x * blockDim.x + threadIdx.x;
  if (i < n) out[i] = (_Float16)in[i];
}

// ---------------------------------------------------------------- GEMM (NT)
// C[M,N] = A[M,K] * B[N,K]^T.  A,B f16 row-major (K contiguous), C = f16/f32.
// Block = 256 threads (8 waves): M-block 256 rows (wave w -> 32 rows),
// N-block 64 cols. B k-slices (64x32 f16) staged global->LDS with
// double-buffered GLOBAL_LOAD_ASYNC_TO_LDS_B128; 8 WMMAs per wave per k-step.
template<typename CT>
__global__ void gemm_nt_wmma(const _Float16* __restrict__ A,
                             const _Float16* __restrict__ B,
                             CT* __restrict__ C,
                             int M, int N, int K) {
  const int tid  = threadIdx.x;
  const int lane = tid & 31;
  const int wave = tid >> 5;
  const int h    = lane >> 4;      // lane half: K sub-block select (ISA 7.12.2)
  const int t16  = lane & 15;
  const int m0   = blockIdx.y * 256 + wave * 32;
  const int n0   = blockIdx.x * 64;

  __shared__ __align__(16) _Float16 s_b[2][64][32];   // 2 x 4KB k-slices of B
  const unsigned lds_base = (unsigned)(uintptr_t)(&s_b[0][0][0]);

  // each thread owns one 16B chunk of the 64x32 B slice
  const int brow   = tid >> 2;         // 0..63  (B row within n-block)
  const int bchunk = tid & 3;          // 0..3   (16B chunk within 32-elem row)
  const _Float16* bgl = B + (size_t)(n0 + brow) * K + bchunk * 8;
  const unsigned lds_dst = lds_base + (unsigned)(brow * 64 + bchunk * 16);

  v8f acc[2][4];
#pragma unroll
  for (int u = 0; u < 2; u++)
#pragma unroll
    for (int t = 0; t < 4; t++) acc[u][t] = (v8f)0.0f;

  const v8h* arow0 = reinterpret_cast<const v8h*>(A + (size_t)(m0 + t16) * K);
  const v8h* arow1 = reinterpret_cast<const v8h*>(A + (size_t)(m0 + 16 + t16) * K);

  // prime buffer 0 with k-slice 0
  async_ld_b128(lds_dst, bgl);
  wait_async0();
  __syncthreads();

  const int nk = K >> 5;
  for (int i = 0; i < nk; i++) {
    const int kk  = i << 5;
    const int buf = i & 1;
    // prefetch next k-slice into the other buffer (overlaps with WMMAs below)
    if (i + 1 < nk) async_ld_b128(lds_dst + (buf ^ 1) * 4096u, bgl + kk + 32);

    // A-frags 16x32 f16: lanes 0-15 K=kk..+15;
    // lanes 16-31 K=kk+16..31
    Frag a0, a1;
    a0.h[0] = arow0[(kk >> 3) + 2 * h];  a0.h[1] = arow0[(kk >> 3) + 2 * h + 1];
    a1.h[0] = arow1[(kk >> 3) + 2 * h];  a1.h[1] = arow1[(kk >> 3) + 2 * h + 1];

#pragma unroll
    for (int t = 0; t < 4; t++) {
      // B-frag 32x16 from LDS: lane (h,t16) holds col n0+16t+t16,
      // rows kk+16h .. kk+16h+15
      const v8h* bl = reinterpret_cast<const v8h*>(&s_b[buf][t * 16 + t16][16 * h]);
      Frag b; b.h[0] = bl[0]; b.h[1] = bl[1];
      acc[0][t] = __builtin_amdgcn_wmma_f32_16x16x32_f16(
          false, a0.v, false, b.v, (short)0, acc[0][t], false, false);
      acc[1][t] = __builtin_amdgcn_wmma_f32_16x16x32_f16(
          false, a1.v, false, b.v, (short)0, acc[1][t], false, false);
    }
    // next buffer landed + everyone done reading current buffer
    if (i + 1 < nk) wait_async0();
    __syncthreads();
  }

  // C/D layout: VGPR r, lane half h -> row +r+8h, col 16t+t16
#pragma unroll
  for (int u = 0; u < 2; u++)
#pragma unroll
    for (int t = 0; t < 4; t++)
#pragma unroll
      for (int r = 0; r < 8; r++)
        C[(size_t)(m0 + u * 16 + r + 8 * h) * N + n0 + t * 16 + t16] =
            (CT)acc[u][t][r];
}

// ---------------------------------------------------------------- RoPE + pack
// in : [S][nh*64] f16 (projection output), out: [nh][S][64] f16 with RoPE.
__global__ void rope_pack_kernel(const _Float16* __restrict__ in,
                                 _Float16* __restrict__ out, int nh) {
  int idx = blockIdx.x * blockDim.x + threadIdx.x;
  int total = S_LEN * nh * HD;
  if (idx >= total) return;
  int d  = idx & 63;
  int hh = (idx >> 6) % nh;
  int s  = idx / (nh * HD);
  int i  = d & 31;
  float invf = __powf(10000.0f, -(float)(2 * i) * (1.0f / 64.0f));
  float ang  = (float)s * invf;
  float c = __cosf(ang), sn = __sinf(ang);
  size_t base = (size_t)s * (nh * HD) + hh * HD;
  float x  = (float)in[base + d];
  float xr = (float)in[base + ((d < 32) ? d + 32 : d - 32)];
  float rot = (d < 32) ? -xr : xr;
  out[((size_t)hh * S_LEN + s) * HD + d] = (_Float16)(x * c + rot * sn);
}

// in : [S][512] f16, out: Vt [8][64][S] f16 (transposed so PV B-frags are contiguous)
__global__ void pack_vt_kernel(const _Float16* __restrict__ in,
                               _Float16* __restrict__ out) {
  int idx = blockIdx.x * blockDim.x + threadIdx.x;
  int total = S_LEN * KVD;
  if (idx >= total) return;
  int d  = idx & 63;
  int hh = (idx >> 6) % NKV;
  int s  = idx / KVD;
  out[((size_t)hh * HD + d) * S_LEN + s] = in[idx];
}

// ---------------------------------------------------------------- attention
// Q: [32][S][64], Kh: [8][S][64], Vt: [8][64][S]; O: [S][2048] f16.
// One wave (32 threads) per (16-query tile, head). Flash-style online softmax,
// key blocks of 32, causal mask, softmax staged through LDS.
__global__ void attn_kernel(const _Float16* __restrict__ Q,
                            const _Float16* __restrict__ Kh,
                            const _Float16* __restrict__ Vt,
                            _Float16* __restrict__ O) {
  const int lane = threadIdx.x;
  const int h    = lane >> 4;
  const int t16  = lane & 15;
  const int i0   = blockIdx.x * 16;
  const int head = blockIdx.y;
  const int kvh  = head >> 2;       // 4 query heads per kv head

  __shared__ __align__(16) float    s_sc[16][32];
  __shared__ __align__(16) _Float16 s_p[16][32];
  __shared__ float s_alpha[16];
  __shared__ float s_l[16];

  // Q fragments for the 16x64 tile: two K-steps of 32
  Frag qf[2];
  const v8h* qrow = reinterpret_cast<const v8h*>(
      Q + ((size_t)head * S_LEN + i0 + t16) * HD);
#pragma unroll
  for (int ks = 0; ks < 2; ks++) {
    qf[ks].h[0] = qrow[ks * 4 + 2 * h];
    qf[ks].h[1] = qrow[ks * 4 + 2 * h + 1];
  }

  v8f acc_o[4];
#pragma unroll
  for (int t = 0; t < 4; t++) acc_o[t] = (v8f)0.0f;
  float m_i = -3.0e38f, l_i = 0.0f;
  const float scale = 0.125f;   // 1/sqrt(64)

  const int jend = (i0 + 16 + 31) >> 5;
  for (int j = 0; j < jend; j++) {
    const int koff = j * 32;
    // ---- scores = Q (16x64) @ K^T (64x32): 2 n-tiles x 2 k-steps
    v8f sc[2];
    sc[0] = (v8f)0.0f; sc[1] = (v8f)0.0f;
#pragma unroll
    for (int nt = 0; nt < 2; nt++) {
      const _Float16* kb =
          Kh + ((size_t)kvh * S_LEN + koff + nt * 16 + t16) * HD;
#pragma unroll
      for (int ks = 0; ks < 2; ks++) {
        const v8h* brow = reinterpret_cast<const v8h*>(kb + ks * 32 + 16 * h);
        Frag b; b.h[0] = brow[0]; b.h[1] = brow[1];
        sc[nt] = __builtin_amdgcn_wmma_f32_16x16x32_f16(
            false, qf[ks].v, false, b.v, (short)0, sc[nt], false, false);
      }
    }
    __syncthreads();
#pragma unroll
    for (int nt = 0; nt < 2; nt++)
#pragma unroll
      for (int r = 0; r < 8; r++)
        s_sc[r + 8 * h][nt * 16 + t16] = sc[nt][r];
    __syncthreads();

    // ---- online softmax: lane m (< 16) owns query row m
    if (lane < 16) {
      const int row = lane, qi = i0 + row;
      float mold = m_i, mx = mold;
      float tv[32];
#pragma unroll
      for (int c = 0; c < 32; c++) {
        float v = s_sc[row][c] * scale;
        if (koff + c > qi) v = -3.0e38f;   // causal mask
        tv[c] = v;
        mx = fmaxf(mx, v);
      }
      float alpha = __expf(mold - mx);
      float sum = 0.0f;
#pragma unroll
      for (int c = 0; c < 32; c++) {
        float p = __expf(tv[c] - mx);
        s_p[row][c] = (_Float16)p;
        sum += p;
      }
      l_i = l_i * alpha + sum;
      m_i = mx;
      s_alpha[row] = alpha;
    }
    __syncthreads();

    // ---- rescale O accumulators by per-row alpha
#pragma unroll
    for (int t = 0; t < 4; t++)
#pragma unroll
      for (int r = 0; r < 8; r++)
        acc_o[t][r] *= s_alpha[r + 8 * h];

    // ---- O += P (16x32) @ V (32x64): re-read P from LDS in A-frag layout
    Frag pa;
    const v8h* pp = reinterpret_cast<const v8h*>(&s_p[t16][0]);
    pa.h[0] = pp[2 * h];
    pa.h[1] = pp[2 * h + 1];
#pragma unroll
    for (int t = 0; t < 4; t++) {
      const v8h* vrow = reinterpret_cast<const v8h*>(
          Vt + ((size_t)kvh * HD + t * 16 + t16) * S_LEN + koff + 16 * h);
      Frag b; b.h[0] = vrow[0]; b.h[1] = vrow[1];
      acc_o[t] = __builtin_amdgcn_wmma_f32_16x16x32_f16(
          false, pa.v, false, b.v, (short)0, acc_o[t], false, false);
    }
    __syncthreads();
  }

  if (lane < 16) s_l[lane] = l_i;
  __syncthreads();
#pragma unroll
  for (int t = 0; t < 4; t++)
#pragma unroll
    for (int r = 0; r < 8; r++) {
      int row = r + 8 * h;
      float inv = 1.0f / s_l[row];
      O[(size_t)(i0 + row) * DIM + head * HD + t * 16 + t16] =
          (_Float16)(acc_o[t][r] * inv);
    }
}

// ---------------------------------------------------------------- launch
extern "C" void kernel_launch(void* const* d_in, const int* in_sizes, int n_in,
                              void* d_out, int out_size, void* d_ws, size_t ws_size,
                              hipStream_t stream) {
  const float* x  = (const float*)d_in[0];
  const float* wq = (const float*)d_in[1];
  const float* wk = (const float*)d_in[2];
  const float* wv = (const float*)d_in[3];
  const float* wo = (const float*)d_in[4];
  float* out = (float*)d_out;

  char* ws = (char*)d_ws;
  size_t off = 0;
  auto alloc = [&](size_t bytes) -> void* {
    void* p = ws + off;
    off += (bytes + 255) & ~(size_t)255;
    return p;
  };
  const size_t SD  = (size_t)S_LEN * DIM;   // 4M elems
  const size_t SKV = (size_t)S_LEN * KVD;   // 1M elems

  _Float16* xh   = (_Float16*)alloc(SD * 2);
  _Float16* wqh  = (_Float16*)alloc(SD * 2);
  _Float16* wkh  = (_Float16*)alloc(SKV * 2);
  _Float16* wvh  = (_Float16*)alloc(SKV * 2);
  _Float16* woh  = (_Float16*)alloc(SD * 2);
  _Float16* q16  = (_Float16*)alloc(SD * 2);
  _Float16* k16  = (_Float16*)alloc(SKV * 2);
  _Float16* v16  = (_Float16*)alloc(SKV * 2);
  _Float16* Qh   = (_Float16*)alloc(SD * 2);
  _Float16* Khp  = (_Float16*)alloc(SKV * 2);
  _Float16* Vt   = (_Float16*)alloc(SKV * 2);
  _Float16* attn = (_Float16*)alloc(SD * 2);
  (void)ws_size; (void)in_sizes; (void)n_in; (void)out_size;

  const int T = 256;
  f32_to_f16_kernel<<<(int)((SD  + T - 1) / T), T, 0, stream>>>(x,  xh,  (int)SD);
  f32_to_f16_kernel<<<(int)((SD  + T - 1) / T), T, 0, stream>>>(wq, wqh, (int)SD);
  f32_to_f16_kernel<<<(int)((SKV + T - 1) / T), T, 0, stream>>>(wk, wkh, (int)SKV);
  f32_to_f16_kernel<<<(int)((SKV + T - 1) / T), T, 0, stream>>>(wv, wvh, (int)SKV);
  f32_to_f16_kernel<<<(int)((SD  + T - 1) / T), T, 0, stream>>>(wo, woh, (int)SD);

  dim3 gq(DIM / 64, S_LEN / 256);
  gemm_nt_wmma<_Float16><<<gq, 256, 0, stream>>>(xh, wqh, q16, S_LEN, DIM, DIM);
  dim3 gkv(KVD / 64, S_LEN / 256);
  gemm_nt_wmma<_Float16><<<gkv, 256, 0, stream>>>(xh, wkh, k16, S_LEN, KVD, DIM);
  gemm_nt_wmma<_Float16><<<gkv, 256, 0, stream>>>(xh, wvh, v16, S_LEN, KVD, DIM);

  rope_pack_kernel<<<(int)((SD  + T - 1) / T), T, 0, stream>>>(q16, Qh, NH);
  rope_pack_kernel<<<(int)((SKV + T - 1) / T), T, 0, stream>>>(k16, Khp, NKV);
  pack_vt_kernel<<<(int)((SKV + T - 1) / T), T, 0, stream>>>(v16, Vt);

  dim3 ga(S_LEN / 16, NH);
  attn_kernel<<<ga, 32, 0, stream>>>(Qh, Khp, Vt, attn);

  dim3 go(DIM / 64, S_LEN / 256);
  gemm_nt_wmma<float><<<go, 256, 0, stream>>>(attn, woh, out, S_LEN, DIM, DIM);
}
